// RandomForestMLP_34986803593499
// MI455X (gfx1250) — matrix operand order repaired and checked by reading, not compiled, not run on hardware
//
#include <hip/hip_runtime.h>

// ---------------------------------------------------------------------------
// Fused 16-tree 3-layer MLP forward for MI455X (gfx1250, wave32, WMMA).
//
//   h1 = relu(x @ W1[t] + b1[t])      [64-row tile, K=512,  N=1024]
//   h2 = relu(h1 @ W2[t] + b2[t])     [K=1024, N=1024]
//   out += (h2 @ W3[t] + b3[t]) / 16  [K=1024, N=256], accumulated in regs
//
// Weights are pre-converted fp32->f16 and pre-transposed to [t][N][K] in d_ws
// so the in-kernel weight stage is contiguous 64B rows. All intermediates live
// in LDS; weights stream from L2 (112MB fp32 -> 56MB f16 resident in 192MB L2)
// via the CDNA5 async global->LDS path (no VGPR round trip, ASYNCcnt-tracked).
// ---------------------------------------------------------------------------

typedef __attribute__((ext_vector_type(16))) _Float16 v16h;
typedef __attribute__((ext_vector_type(8)))  float    v8f;

#define T_TREES 16
#define NB      8192
#define DIN     512
#define DH      1024
#define DOUT    256

#define BT       64                 // batch rows per block
#define NCHUNK   512                // N columns staged per pass
#define KSTEP    32                 // K per WMMA
#define NWAVES   16
#define NTHREADS (NWAVES * 32)

// LDS strides in f16 elements; +8 padding keeps 16-lane b128 loads conflict-free
#define XSTR   (DIN + 8)            // 520
#define HSTR   (DH + 8)             // 1032
#define WKSTR  (KSTEP + 8)          // 40

#define REGA_OFF  0                 // x (layer1 in), then h2 (layer2 out / layer3 in)
#define REGB_OFF  (BT * HSTR)       // h1
#define WST_OFF   (2 * BT * HSTR)   // weight stage [NCHUNK][WKSTR]
#define LDS_ELEMS (WST_OFF + NCHUNK * WKSTR)
#define LDS_BYTES (LDS_ELEMS * 2)

static_assert(LDS_BYTES <= 320 * 1024, "exceeds 320KB WGP LDS");

// ---------------------------------------------------------------------------
// Prologue: W[t][K][N] fp32 -> Wt[t][N][K] f16 (tiled transpose through LDS)
// ---------------------------------------------------------------------------
__global__ void __launch_bounds__(256)
wconv_transpose_kernel(const float* __restrict__ W, _Float16* __restrict__ Wt,
                       int K, int N) {
    __shared__ float tile[32][33];
    const int t  = blockIdx.z;
    const int k0 = blockIdx.x * 32;
    const int n0 = blockIdx.y * 32;
    const float*    Wp  = W  + (size_t)t * K * N;
    _Float16*       Wtp = Wt + (size_t)t * K * N;
    const int lx = threadIdx.x;   // 0..31
    const int ly = threadIdx.y;   // 0..7
#pragma unroll
    for (int i = 0; i < 32; i += 8)
        tile[ly + i][lx] = Wp[(size_t)(k0 + ly + i) * N + (n0 + lx)];
    __syncthreads();
#pragma unroll
    for (int i = 0; i < 32; i += 8)
        Wtp[(size_t)(n0 + ly + i) * K + (k0 + lx)] = (_Float16)tile[lx][ly + i];
}

// ---------------------------------------------------------------------------
// Fragment loaders (ISA 16-bit A / B VGPR layouts, wave32)
// ---------------------------------------------------------------------------
__device__ __forceinline__ v16h load_afrag(const _Float16* rowptr) {
    // lanes 0-15: K {0..7,16..23}; lanes 16-31: K {8..15,24..31} (caller adds koff)
    v16h a;
    ((uint4*)&a)[0] = *(const uint4*)(rowptr);
    ((uint4*)&a)[1] = *(const uint4*)(rowptr + 16);
    return a;
}
__device__ __forceinline__ v16h load_bfrag(const _Float16* colptr) {
    // per-lane: one N column; lanes 0-15: K {0..7,16..23}; lanes 16-31: K {8..15,24..31} (caller adds 8*(lane>>4))
    v16h b;
    ((uint4*)&b)[0] = *(const uint4*)(colptr);
    ((uint4*)&b)[1] = *(const uint4*)(colptr + 16);
    return b;
}

__device__ __forceinline__ v8f zero8() {
    v8f z = {0.f, 0.f, 0.f, 0.f, 0.f, 0.f, 0.f, 0.f};
    return z;
}

// ---------------------------------------------------------------------------
// Async weight staging: one [nrows x 32] K-slice of transposed f16 weights,
// copied global->LDS with GLOBAL_LOAD_ASYNC_TO_LDS_B128 (16B per lane, no
// destination VGPRs, tracked by ASYNCcnt). Caller brackets with barriers.
// ---------------------------------------------------------------------------
__device__ __forceinline__ void async_stage_w(const _Float16* __restrict__ wt, int Ktot,
                                              int nbase, int nrows, int k0,
                                              _Float16* __restrict__ wst, int tid) {
    const int nchunks = nrows * 4;                 // 16B (8 x f16) chunks
    for (int c = tid; c < nchunks; c += NTHREADS) {
        const int r    = c >> 2;
        const int part = (c & 3) << 3;             // f16 offset within 64B row
        const _Float16* gp = wt + (size_t)(nbase + r) * Ktot + k0 + part;
        const unsigned int lds_addr =
            (unsigned int)(uintptr_t)(wst + (size_t)r * WKSTR + part);
        asm volatile("global_load_async_to_lds_b128 %0, %1, off"
                     :: "v"(lds_addr), "v"(gp)
                     : "memory");
    }
    asm volatile("s_wait_asynccnt 0x0" ::: "memory");
}

// One dense layer: hout = act(hin @ W + b), 64 rows, chunked over N.
// Wave w owns col tiles {2w, 2w+1} of each 512-wide chunk, all 4 row tiles.
__device__ __forceinline__ void layer_pass(const _Float16* __restrict__ hin, int inStride, int Ktot,
                                           const _Float16* __restrict__ wtg, int Nout,
                                           const float* __restrict__ bias,
                                           _Float16* __restrict__ hout, int outStride,
                                           _Float16* __restrict__ wst,
                                           int tid, int lane, int wave) {
    const int m     = lane & 15;
    const int koff8 = (lane >> 4) << 3;   // A-frag K offset
    const int kk16  = (lane >> 4) << 4;   // unused (B-frag also uses koff8)
    const int nchunks = Nout / NCHUNK;
    for (int ch = 0; ch < nchunks; ++ch) {
        const int nbase = ch * NCHUNK;
        v8f acc[4][2];
#pragma unroll
        for (int r = 0; r < 4; ++r)
#pragma unroll
            for (int c = 0; c < 2; ++c) acc[r][c] = zero8();

        for (int k0 = 0; k0 < Ktot; k0 += KSTEP) {
            __syncthreads();                          // wstage readers done
            async_stage_w(wtg, Ktot, nbase, NCHUNK, k0, wst, tid);
            __syncthreads();                          // wstage published
            v16h a[4], b[2];
#pragma unroll
            for (int r = 0; r < 4; ++r)
                a[r] = load_afrag(hin + (size_t)(r * 16 + m) * inStride + k0 + koff8);
#pragma unroll
            for (int c = 0; c < 2; ++c) {
                const int ct = wave * 2 + c;          // col tile in chunk (0..31)
                b[c] = load_bfrag(wst + (size_t)(ct * 16 + m) * WKSTR + koff8);
            }
#pragma unroll
            for (int r = 0; r < 4; ++r)
#pragma unroll
                for (int c = 0; c < 2; ++c)
                    acc[r][c] = __builtin_amdgcn_wmma_f32_16x16x32_f16(
                        false, a[r], false, b[c], (short)0, acc[r][c], false, false);
        }

        // epilogue: bias + relu + f16 store to LDS
        const int mbase = (lane >> 4) << 3;
#pragma unroll
        for (int c = 0; c < 2; ++c) {
            const int ctg  = nbase + (wave * 2 + c) * 16;   // global col base of tile
            const float bv = bias[ctg + m];
#pragma unroll
            for (int r = 0; r < 4; ++r) {
                _Float16* dp = hout + (size_t)(r * 16 + mbase) * outStride + ctg + m;
#pragma unroll
                for (int v = 0; v < 8; ++v) {
                    float f = fmaxf(acc[r][c][v] + bv, 0.f);
                    dp[(size_t)v * outStride] = (_Float16)f;
                }
            }
        }
    }
}

// ---------------------------------------------------------------------------
// Main fused kernel: one block = 64 batch rows, loops over all 16 trees.
// ---------------------------------------------------------------------------
__global__ void __launch_bounds__(NTHREADS, 1)
fused_forest_mlp(const float* __restrict__ x,
                 const _Float16* __restrict__ w1t,
                 const _Float16* __restrict__ w2t,
                 const _Float16* __restrict__ w3t,
                 const float* __restrict__ b1,
                 const float* __restrict__ b2,
                 const float* __restrict__ b3,
                 float* __restrict__ out) {
    extern __shared__ _Float16 lds[];
    _Float16* regA = lds + REGA_OFF;
    _Float16* regB = lds + REGB_OFF;
    _Float16* wst  = lds + WST_OFF;

    const int tid     = threadIdx.x;
    const int lane    = tid & 31;
    const int wave    = tid >> 5;
    const int rowbase = blockIdx.x * BT;
    const int m       = lane & 15;
    const int koff8   = (lane >> 4) << 3;
    const int kk16    = (lane >> 4) << 4;

    // layer-3 accumulators persist across trees: wave owns col tile c=wave (N=256)
    v8f acc3[4];
#pragma unroll
    for (int r = 0; r < 4; ++r) acc3[r] = zero8();

    for (int t = 0; t < T_TREES; ++t) {
        __syncthreads();   // regA (h2 of previous tree) fully consumed

        // ---- stage x tile -> regA as f16 (rows BT, stride XSTR) ----
        for (int idx = tid; idx < BT * (DIN / 4); idx += NTHREADS) {
            const int row = idx >> 7;          // DIN/4 == 128
            const int c4  = idx & 127;
            const float4 v = ((const float4*)x)[(((size_t)(rowbase + row) * DIN) >> 2) + c4];
            _Float16 h4[4] = {(_Float16)v.x, (_Float16)v.y, (_Float16)v.z, (_Float16)v.w};
            *(uint2*)(regA + (size_t)row * XSTR + (c4 << 2)) = *(uint2*)h4;
        }
        // (layer_pass's first barrier orders these writes before reads)

        // ---- layer 1: regA(x) -> regB(h1) ----
        layer_pass(regA, XSTR, DIN, w1t + (size_t)t * DIN * DH, DH,
                   b1 + (size_t)t * DH, regB, HSTR, wst, tid, lane, wave);
        // ---- layer 2: regB(h1) -> regA(h2) ----
        layer_pass(regB, HSTR, DH, w2t + (size_t)t * DH * DH, DH,
                   b2 + (size_t)t * DH, regA, HSTR, wst, tid, lane, wave);

        // ---- layer 3: regA(h2) @ W3 -> acc3 (no store; bias folded at the end) ----
        for (int k0 = 0; k0 < DH; k0 += KSTEP) {
            __syncthreads();
            async_stage_w(w3t + (size_t)t * DH * DOUT, DH, 0, DOUT, k0, wst, tid);
            __syncthreads();
            v16h a[4];
#pragma unroll
            for (int r = 0; r < 4; ++r)
                a[r] = load_afrag(regA + (size_t)(r * 16 + m) * HSTR + k0 + koff8);
            const v16h b = load_bfrag(wst + (size_t)(wave * 16 + m) * WKSTR + koff8);
#pragma unroll
            for (int r = 0; r < 4; ++r)
                acc3[r] = __builtin_amdgcn_wmma_f32_16x16x32_f16(
                    false, a[r], false, b, (short)0, acc3[r], false, false);
        }
    }

    // ---- final: out = acc3/T + mean_t(b3) ----
    {
        const int  colg  = wave * 16 + m;
        const int  mbase = (lane >> 4) << 3;
        float bsum = 0.f;
#pragma unroll
        for (int t = 0; t < T_TREES; ++t) bsum += b3[(size_t)t * DOUT + colg];
        const float inv = 1.f / (float)T_TREES;
        bsum *= inv;
#pragma unroll
        for (int r = 0; r < 4; ++r) {
            float* op = out + (size_t)(rowbase + r * 16 + mbase) * DOUT + colg;
#pragma unroll
            for (int v = 0; v < 8; ++v)
                op[(size_t)v * DOUT] = acc3[r][v] * inv + bsum;
        }
    }
}

// ---------------------------------------------------------------------------
// Launcher
// ---------------------------------------------------------------------------
extern "C" void kernel_launch(void* const* d_in, const int* in_sizes, int n_in,
                              void* d_out, int out_size, void* d_ws, size_t ws_size,
                              hipStream_t stream) {
    (void)in_sizes; (void)n_in; (void)out_size; (void)ws_size;
    const float* x  = (const float*)d_in[0];
    const float* W1 = (const float*)d_in[1];
    const float* b1 = (const float*)d_in[2];
    const float* W2 = (const float*)d_in[3];
    const float* b2 = (const float*)d_in[4];
    const float* W3 = (const float*)d_in[5];
    const float* b3 = (const float*)d_in[6];
    float* out = (float*)d_out;

    _Float16* w1t = (_Float16*)d_ws;                               // [T][DH][DIN]
    _Float16* w2t = w1t + (size_t)T_TREES * DIN * DH;              // [T][DH][DH]
    _Float16* w3t = w2t + (size_t)T_TREES * DH * DH;               // [T][DOUT][DH]

    const dim3 tb(32, 8);
    wconv_transpose_kernel<<<dim3(DIN / 32, DH / 32, T_TREES), tb, 0, stream>>>(W1, w1t, DIN, DH);
    wconv_transpose_kernel<<<dim3(DH / 32, DH / 32, T_TREES),  tb, 0, stream>>>(W2, w2t, DH, DH);
    wconv_transpose_kernel<<<dim3(DH / 32, DOUT / 32, T_TREES), tb, 0, stream>>>(W3, w3t, DH, DOUT);

    hipFuncSetAttribute((const void*)fused_forest_mlp,
                        hipFuncAttributeMaxDynamicSharedMemorySize, LDS_BYTES);
    fused_forest_mlp<<<dim3(NB / BT), dim3(NTHREADS), LDS_BYTES, stream>>>(
        x, w1t, w2t, w3t, b1, b2, b3, out);
}
